// structured_ouroboros_69492570849415
// MI455X (gfx1250) — matrix operand order repaired and checked by reading, not compiled, not run on hardware
//
#include <hip/hip_runtime.h>

// ---------------------------------------------------------------------------
// structured_ouroboros for MI455X (gfx1250, wave32, WMMA bf16, async-to-LDS)
// ---------------------------------------------------------------------------
#define B_SZ   2
#define L_SEQ  1024
#define T_TOK  (B_SZ * L_SEQ)      // 2048 tokens
#define D_DATA 8
#define D_CTRL 256
#define POLY   40
#define N_LAYERS 2
#define D_STATE 16
#define D_CONV 4
#define D_INNER 256
#define DT_RANK 16
#define ZOUT   (B_SZ * (L_SEQ - 1) * 2 * D_DATA)   // 32736 floats of zdot output

typedef __attribute__((ext_vector_type(16))) __bf16 v16bf;
typedef __attribute__((ext_vector_type(8)))  float  v8f;
typedef unsigned int uiv4 __attribute__((ext_vector_type(4)));

union FragCast { uiv4 u[2]; v16bf v; };

static __device__ __forceinline__ unsigned short f2bf(float f) {
  unsigned int u = __float_as_uint(f);
  u = (u + 0x7FFFu + ((u >> 16) & 1u)) >> 16;   // round-to-nearest-even
  return (unsigned short)u;
}

static __device__ __forceinline__ float silu(float x) {
  return x / (1.f + __expf(-x));
}

// ---------------------------------------------------------------------------
// Weight conversion f32 -> bf16 (runs every launch; deterministic)
// ---------------------------------------------------------------------------
__global__ void cvt_bf16_kernel(const float* __restrict__ src,
                                unsigned short* __restrict__ dst, int n) {
  for (int i = blockIdx.x * blockDim.x + threadIdx.x; i < n;
       i += gridDim.x * blockDim.x)
    dst[i] = f2bf(src[i]);
}

// ---------------------------------------------------------------------------
// prep: xdot, power matrix pm[t][16][40], ctrl_in (flipped) = u buffer
// ---------------------------------------------------------------------------
__global__ void prep_kernel(const float* __restrict__ x, const float* __restrict__ y,
                            const float* __restrict__ cw, const float* __restrict__ cb,
                            float* __restrict__ pm, float* __restrict__ u) {
  int t = blockIdx.x;                   // 0..2047 (original order)
  int b = t >> 10, l = t & 1023;
  __shared__ float c16[16];             // [xdot, y]
  __shared__ float z16[16];             // [x, xdot]
  int e = threadIdx.x;
  if (e < 8) {
    float xv = x[t * 8 + e], yv = y[t * 8 + e];
    float xd = yv - xv;
    z16[e] = xv; z16[8 + e] = xd;
    c16[e] = xd; c16[8 + e] = yv;
  }
  __syncthreads();
  if (e < 16) {                         // cumprod powers z^0..z^39
    float zv = z16[e], p = 1.f;
    float* pmt = pm + ((size_t)t * 16 + e) * POLY;
    for (int j = 0; j < POLY; ++j) { pmt[j] = p; p *= zv; }
  }
  int tf = b * L_SEQ + (L_SEQ - 1 - l); // ctrl_in is flipped along l
  float acc = cb[e];
#pragma unroll
  for (int i = 0; i < 16; ++i) acc += cw[e * 16 + i] * c16[i];
  u[(size_t)tf * D_CTRL + e] = acc;
}

// ---------------------------------------------------------------------------
// RMSNorm -> bf16 activations
// ---------------------------------------------------------------------------
__global__ void rmsnorm_kernel(const float* __restrict__ u,
                               const float* __restrict__ norm_w,
                               unsigned short* __restrict__ r_bf, int layer) {
  __shared__ float red[8];
  int tf = blockIdx.x, e = threadIdx.x;
  float v = u[(size_t)tf * D_CTRL + e];
  float sq = v * v;
#pragma unroll
  for (int m = 1; m < 32; m <<= 1) sq += __shfl_xor(sq, m);
  if ((e & 31) == 0) red[e >> 5] = sq;
  __syncthreads();
  float s = 0.f;
#pragma unroll
  for (int i = 0; i < 8; ++i) s += red[i];
  float scale = rsqrtf(s * (1.f / 256.f) + 1e-5f);
  r_bf[(size_t)tf * D_CTRL + e] = f2bf(v * scale * norm_w[layer * D_CTRL + e]);
}

// ---------------------------------------------------------------------------
// Generic GEMM  C[M,N](f32) = A[M,K](bf16) @ W[N,K](bf16)^T (+bias)(+resid)
// ---------------------------------------------------------------------------
__global__ void gemm_bf16_wmma(const unsigned short* __restrict__ A,
                               const unsigned short* __restrict__ W,
                               const float* __restrict__ bias,
                               const float* __restrict__ resid,
                               float* __restrict__ C, int N, int K) {
  int mt = blockIdx.x, nt = blockIdx.y;
  int lane = threadIdx.x & 31;
  int r = lane & 15, h = lane >> 4;
  const unsigned short* arow = A + (size_t)(mt * 16 + r) * K;
  const unsigned short* wrow = W + (size_t)(nt * 16 + r) * K;
  v8f acc = {};
  for (int k0 = 0; k0 < K; k0 += 32) {
    FragCast fa, fb;
    fa.u[0] = *(const uiv4*)(arow + k0 + h * 8);         // A: K {0..7|8..15}
    fa.u[1] = *(const uiv4*)(arow + k0 + 16 + h * 8);    //    K {16..23|24..31}
    fb.u[0] = *(const uiv4*)(wrow + k0 + h * 8);         // B: col=r, K {0..7|8..15}
    fb.u[1] = *(const uiv4*)(wrow + k0 + 16 + h * 8);    //    K {16..23|24..31}
    acc = __builtin_amdgcn_wmma_f32_16x16x32_bf16(false, fa.v, false, fb.v,
                                                  (short)0, acc, false, false);
  }
  int n = nt * 16 + r;
  int mbase = mt * 16 + h * 8;
  float bv = bias ? bias[n] : 0.f;
#pragma unroll
  for (int vi = 0; vi < 8; ++vi) {
    size_t idx = (size_t)(mbase + vi) * N + n;
    float val = acc[vi] + bv;
    if (resid) val += resid[idx];
    C[idx] = val;
  }
}

// ---------------------------------------------------------------------------
// Causal conv (K=4) + SiLU
// ---------------------------------------------------------------------------
__global__ void conv_kernel(const float* __restrict__ xz,
                            const float* __restrict__ cw, const float* __restrict__ cb,
                            float* __restrict__ xc, unsigned short* __restrict__ xc_bf,
                            int layer) {
  int idx = blockIdx.x * 256 + threadIdx.x;   // tf*256 + d
  int tf = idx >> 8, d = idx & 255;
  int lf = tf & 1023, bbase = tf - lf;
  float acc = cb[layer * D_INNER + d];
#pragma unroll
  for (int k = 0; k < D_CONV; ++k) {
    int ls = lf - (D_CONV - 1) + k;
    if (ls >= 0)
      acc += xz[(size_t)(bbase + ls) * (2 * D_INNER) + d] *
             cw[(size_t)(layer * D_INNER + d) * D_CONV + k];
  }
  float s = silu(acc);
  xc[idx] = s;
  xc_bf[idx] = f2bf(s);
}

// ---------------------------------------------------------------------------
// delta = softplus(dbc[:, :16] @ dt_w^T + dt_b)
// ---------------------------------------------------------------------------
__global__ void dt_kernel(const float* __restrict__ dbc,
                          const float* __restrict__ dtw, const float* __restrict__ dtb,
                          float* __restrict__ delta, int layer) {
  int idx = blockIdx.x * 256 + threadIdx.x;
  int tf = idx >> 8, d = idx & 255;
  float acc = dtb[layer * D_INNER + d];
#pragma unroll
  for (int rr = 0; rr < DT_RANK; ++rr)
    acc += dbc[(size_t)tf * 48 + rr] * dtw[(size_t)(layer * D_INNER + d) * DT_RANK + rr];
  delta[idx] = (acc > 20.f) ? acc : log1pf(__expf(acc));
}

// ---------------------------------------------------------------------------
// Selective-scan (wave32 shfl reduction over 16 states)
// ---------------------------------------------------------------------------
__global__ void scan_kernel(const float* __restrict__ delta,
                            const float* __restrict__ dbc,
                            const float* __restrict__ xc,
                            const float* __restrict__ xz,
                            const float* __restrict__ A_log,
                            const float* __restrict__ D_skip,
                            unsigned short* __restrict__ yv_bf, int layer) {
  int b = blockIdx.x;
  int d = blockIdx.y * 16 + (threadIdx.x >> 4);
  int s = threadIdx.x & 15;
  float A   = -__expf(A_log[(size_t)(layer * D_INNER + d) * D_STATE + s]);
  float Dsk = D_skip[layer * D_INNER + d];
  float h = 0.f;
  for (int lf = 0; lf < L_SEQ; ++lf) {
    int tf = b * L_SEQ + lf;
    float dt  = delta[(size_t)tf * D_INNER + d];
    float Bm  = dbc[(size_t)tf * 48 + DT_RANK + s];
    float Cm  = dbc[(size_t)tf * 48 + DT_RANK + D_STATE + s];
    float xcv = xc[(size_t)tf * D_INNER + d];
    h = __expf(dt * A) * h + dt * Bm * xcv;
    float p = h * Cm;
    p += __shfl_xor(p, 1); p += __shfl_xor(p, 2);
    p += __shfl_xor(p, 4); p += __shfl_xor(p, 8);
    if (s == 0) {
      float zg = xz[(size_t)tf * (2 * D_INNER) + D_INNER + d];
      float yv = (p + Dsk * xcv) * silu(zg);
      yv_bf[(size_t)tf * D_INNER + d] = f2bf(yv);
    }
  }
}

// ---------------------------------------------------------------------------
// un-flip sp -> state_pred output + bf16 copy for the big GEMM
// ---------------------------------------------------------------------------
__global__ void state_out_kernel(const float* __restrict__ u,
                                 float* __restrict__ out,
                                 unsigned short* __restrict__ sp_bf) {
  int idx = blockIdx.x * 256 + threadIdx.x;   // tf*256 + e
  int tf = idx >> 8, e = idx & 255;
  int b = tf >> 10, lf = tf & 1023;
  int t = b * L_SEQ + (L_SEQ - 1 - lf);
  float v = u[idx];
  out[ZOUT + (size_t)t * D_CTRL + e] = v;
  sp_bf[(size_t)t * D_CTRL + e] = f2bf(v);
}

// ---------------------------------------------------------------------------
// Fused b_net GEMM + polynomial contraction.
//  grid (128 token-tiles, 16 dd), block 256 (8 waves).
//  B tile (16 tokens x 256 bf16 = 8KB) staged ONCE into LDS with
//  global_load_async_to_lds_b128 (ASYNCcnt path), so the 8 waves' 100 row-tile
//  passes read B from LDS instead of re-fetching 800KB through L0.
//  A row-tiles stream from L2 with a one-tile-ahead global_prefetch.
// ---------------------------------------------------------------------------
__global__ void bigpoly_kernel(const unsigned short* __restrict__ spb,
                               const unsigned short* __restrict__ Wb,
                               const float* __restrict__ bb,
                               const float* __restrict__ pm,
                               float* __restrict__ zws) {
  int tt = blockIdx.x;     // token tile 0..127
  int dd = blockIdx.y;     // 0..15
  __shared__ __align__(16) unsigned short bsh[16 * D_CTRL];  // 8KB B tile
  __shared__ float pmA[16 * POLY];   // pm[token, dd, :]
  __shared__ float pmB[16 * POLY];   // pm[token, 15-dd, :]
  __shared__ float zp[8][16];
  int tid = threadIdx.x;

  // ---- async-stage B tile: each thread copies two 16B chunks (8KB total)
  {
    unsigned lbase = (unsigned)(unsigned long long)(&bsh[0]);   // LDS byte offset
    unsigned long long gbase =
        (unsigned long long)(spb + (size_t)tt * 16 * D_CTRL);
#pragma unroll
    for (int rnd = 0; rnd < 2; ++rnd) {
      int chunk = tid + rnd * 256;                  // 16-byte chunk index
      unsigned laddr = lbase + (unsigned)chunk * 16u;
      unsigned long long gaddr = gbase + (unsigned long long)chunk * 16ull;
      asm volatile("global_load_async_to_lds_b128 %0, %1, off"
                   :: "v"(laddr), "v"(gaddr) : "memory");
    }
  }

  // ---- stage pm slabs while the async copy is in flight
  for (int i = tid; i < 16 * POLY; i += 256) {
    int tok = i / POLY, j = i - tok * POLY;
    int t = tt * 16 + tok;
    pmA[i] = pm[((size_t)t * 16 + dd) * POLY + j];
    pmB[i] = pm[((size_t)t * 16 + (15 - dd)) * POLY + j];
  }

  asm volatile("s_wait_asynccnt 0x0" ::: "memory");
  __syncthreads();

  int wave = tid >> 5, lane = tid & 31;
  int r = lane & 15, h = lane >> 4;
  const unsigned short* brow = bsh + r * D_CTRL;   // B fragments from LDS
  float zacc = 0.f;
  for (int rt = wave; rt < 100; rt += 8) {           // 1600 rows per dd
    int rowbase = dd * 1600 + rt * 16;
    const unsigned short* arow = Wb + (size_t)(rowbase + r) * D_CTRL;
    if (rt + 8 < 100)                                // keep L2->L0 stream ahead
      __builtin_prefetch(Wb + (size_t)(rowbase + 128 + r) * D_CTRL, 0, 1);
    v8f acc = {};
#pragma unroll
    for (int k0 = 0; k0 < D_CTRL; k0 += 32) {
      FragCast fa, fb;
      fa.u[0] = *(const uiv4*)(arow + k0 + h * 8);
      fa.u[1] = *(const uiv4*)(arow + k0 + 16 + h * 8);
      fb.u[0] = *(const uiv4*)(brow + k0 + h * 8);
      fb.u[1] = *(const uiv4*)(brow + k0 + 16 + h * 8);
      acc = __builtin_amdgcn_wmma_f32_16x16x32_bf16(false, fa.v, false, fb.v,
                                                    (short)0, acc, false, false);
    }
    // contract this 16x16 bmat tile: rows -> (k,j), cols -> token r
    float partial = 0.f;
    int mb = h * 8;
#pragma unroll
    for (int vi = 0; vi < 8; ++vi) {
      int rr = rt * 16 + mb + vi;          // 0..1599 within dd block
      int k = rr / POLY, j = rr - k * POLY;
      float val = acc[vi] + bb[rowbase + mb + vi];
      partial += val * pmA[r * POLY + k] * pmB[r * POLY + j];
    }
    partial += __shfl_xor(partial, 16);    // fold upper-half rows
    if (h == 0) zacc += partial;
  }
  if (lane < 16) zp[wave][lane] = zacc;
  __syncthreads();
  if (tid < 16) {
    float s = 0.f;
#pragma unroll
    for (int w = 0; w < 8; ++w) s += zp[w][tid];
    zws[(size_t)(tt * 16 + tid) * 16 + dd] = s;
  }
}

// ---------------------------------------------------------------------------
// pack zdot[:, 1:, :] into d_out head
// ---------------------------------------------------------------------------
__global__ void pack_zdot_kernel(const float* __restrict__ zws, float* __restrict__ out) {
  int idx = blockIdx.x * 256 + threadIdx.x;
  if (idx >= ZOUT) return;
  int dd = idx & 15;
  int rest = idx >> 4;
  int l1 = rest % (L_SEQ - 1);
  int b  = rest / (L_SEQ - 1);
  out[idx] = zws[(size_t)(b * L_SEQ + (l1 + 1)) * 16 + dd];
}

// ---------------------------------------------------------------------------
extern "C" void kernel_launch(void* const* d_in, const int* in_sizes, int n_in,
                              void* d_out, int out_size, void* d_ws, size_t ws_size,
                              hipStream_t stream) {
  (void)in_sizes; (void)n_in; (void)out_size; (void)ws_size;
  const float* x        = (const float*)d_in[0];
  const float* y        = (const float*)d_in[1];
  const float* cpw      = (const float*)d_in[2];
  const float* cpb      = (const float*)d_in[3];
  const float* norm_w   = (const float*)d_in[4];
  const float* in_pw    = (const float*)d_in[5];
  const float* conv_w   = (const float*)d_in[6];
  const float* conv_b   = (const float*)d_in[7];
  const float* x_pw     = (const float*)d_in[8];
  const float* dt_pw    = (const float*)d_in[9];
  const float* dt_pb    = (const float*)d_in[10];
  const float* A_log    = (const float*)d_in[11];
  const float* D_skip   = (const float*)d_in[12];
  const float* out_pw   = (const float*)d_in[13];
  const float* bnet_w   = (const float*)d_in[14];
  const float* bnet_b   = (const float*)d_in[15];
  float* out = (float*)d_out;
  char*  ws  = (char*)d_ws;

  size_t off = 0;
  auto alloc = [&](size_t bytes) -> char* {
    char* p = ws + off;
    off = (off + bytes + 255) & ~(size_t)255;
    return p;
  };
  unsigned short* inpw_bf = (unsigned short*)alloc((size_t)N_LAYERS * 2 * D_INNER * D_CTRL * 2);
  unsigned short* xpw_bf  = (unsigned short*)alloc((size_t)N_LAYERS * 48 * D_INNER * 2);
  unsigned short* opw_bf  = (unsigned short*)alloc((size_t)N_LAYERS * D_CTRL * D_INNER * 2);
  unsigned short* bnet_bf = (unsigned short*)alloc((size_t)25600 * D_CTRL * 2);
  float* pm     = (float*)alloc((size_t)T_TOK * 16 * POLY * 4);
  float* u      = (float*)alloc((size_t)T_TOK * D_CTRL * 4);
  unsigned short* r_bf  = (unsigned short*)alloc((size_t)T_TOK * D_CTRL * 2);
  float* xz     = (float*)alloc((size_t)T_TOK * 2 * D_INNER * 4);
  float* xc     = (float*)alloc((size_t)T_TOK * D_INNER * 4);
  unsigned short* xc_bf = (unsigned short*)alloc((size_t)T_TOK * D_INNER * 2);
  float* dbc    = (float*)alloc((size_t)T_TOK * 48 * 4);
  float* delta  = (float*)alloc((size_t)T_TOK * D_INNER * 4);
  unsigned short* yv_bf = (unsigned short*)alloc((size_t)T_TOK * D_INNER * 2);
  unsigned short* sp_bf = (unsigned short*)alloc((size_t)T_TOK * D_CTRL * 2);
  float* zws    = (float*)alloc((size_t)T_TOK * 16 * 4);

  // weight conversions
  cvt_bf16_kernel<<<512, 256, 0, stream>>>(in_pw, inpw_bf, N_LAYERS * 2 * D_INNER * D_CTRL);
  cvt_bf16_kernel<<<96, 256, 0, stream>>>(x_pw, xpw_bf, N_LAYERS * 48 * D_INNER);
  cvt_bf16_kernel<<<256, 256, 0, stream>>>(out_pw, opw_bf, N_LAYERS * D_CTRL * D_INNER);
  cvt_bf16_kernel<<<4096, 256, 0, stream>>>(bnet_w, bnet_bf, 25600 * D_CTRL);

  // prep: pm + ctrl_in (flipped) into u
  prep_kernel<<<T_TOK, 256, 0, stream>>>(x, y, cpw, cpb, pm, u);

  for (int layer = 0; layer < N_LAYERS; ++layer) {
    rmsnorm_kernel<<<T_TOK, 256, 0, stream>>>(u, norm_w, r_bf, layer);

    // xz = r @ in_proj^T   (2048 x 512, K=256)
    gemm_bf16_wmma<<<dim3(T_TOK / 16, (2 * D_INNER) / 16), 32, 0, stream>>>(
        r_bf, inpw_bf + (size_t)layer * 2 * D_INNER * D_CTRL,
        nullptr, nullptr, xz, 2 * D_INNER, D_CTRL);

    conv_kernel<<<(T_TOK * D_INNER) / 256, 256, 0, stream>>>(
        xz, conv_w, conv_b, xc, xc_bf, layer);

    // dbc = xc @ x_proj^T  (2048 x 48, K=256)
    gemm_bf16_wmma<<<dim3(T_TOK / 16, 48 / 16), 32, 0, stream>>>(
        xc_bf, xpw_bf + (size_t)layer * 48 * D_INNER,
        nullptr, nullptr, dbc, 48, D_INNER);

    dt_kernel<<<(T_TOK * D_INNER) / 256, 256, 0, stream>>>(dbc, dt_pw, dt_pb, delta, layer);

    scan_kernel<<<dim3(B_SZ, D_INNER / 16), 256, 0, stream>>>(
        delta, dbc, xc, xz, A_log, D_skip, yv_bf, layer);

    // u += yv @ out_proj^T  (2048 x 256, K=256, residual in-place)
    gemm_bf16_wmma<<<dim3(T_TOK / 16, D_CTRL / 16), 32, 0, stream>>>(
        yv_bf, opw_bf + (size_t)layer * D_CTRL * D_INNER,
        nullptr, u, u, D_CTRL, D_INNER);
  }

  // state_pred output (un-flip) + bf16 for big GEMM
  state_out_kernel<<<(T_TOK * D_CTRL) / 256, 256, 0, stream>>>(u, out, sp_bf);

  // fused b_net GEMM + poly contraction -> zdot workspace
  bigpoly_kernel<<<dim3(T_TOK / 16, 16), 256, 0, stream>>>(sp_bf, bnet_bf, bnet_b, pm, zws);

  // zdot[:, 1:, :] -> d_out head
  pack_zdot_kernel<<<(ZOUT + 255) / 256, 256, 0, stream>>>(zws, out);
}
